// Router_10307921510766
// MI455X (gfx1250) — matrix operand order repaired and checked by reading, not compiled, not run on hardware
//
#include <hip/hip_runtime.h>
#include <hip/hip_bf16.h>
#include <math.h>

typedef __attribute__((ext_vector_type(16))) __bf16 v16bf;
typedef __attribute__((ext_vector_type(8)))  __bf16 v8bf;
typedef __attribute__((ext_vector_type(8)))  float  v8f;
typedef __attribute__((ext_vector_type(4)))  float  v4f;

constexpr int kB = 4;
constexpr int kS = 4096;
constexpr int kD = 4096;
constexpr int kE = 64;
constexpr int kTopK = 8;
constexpr int kRows = kB * kS;          // 16384 tokens
constexpr int kTokPerWave = 16;         // token tile per wave (WMMA N)
constexpr int kWavesPerBlk = 2;
constexpr int kTokPerBlk = kTokPerWave * kWavesPerBlk;   // 32
constexpr int kTileK = 32;              // WMMA bf16 K

// ---------------------------------------------------------------------------
// Prepass: fp32 weights [D][E] -> transposed bf16 [E][D]. Serves the WMMA
// A-fragments as contiguous 16B runs. 1 MB total -> L2-resident everywhere.
// ---------------------------------------------------------------------------
__global__ void cvt_weights_kernel(const float* __restrict__ Wg,
                                   const float* __restrict__ Wn,
                                   __bf16* __restrict__ WgT,
                                   __bf16* __restrict__ WnT) {
    int idx = blockIdx.x * blockDim.x + threadIdx.x;   // over D*E
    if (idx >= kD * kE) return;
    int k = idx >> 6;        // row in [D]
    int e = idx & (kE - 1);  // col in [E]
    WgT[(size_t)e * kD + k] = (__bf16)Wg[idx];
    WnT[(size_t)e * kD + k] = (__bf16)Wn[idx];
}

// Build a v16bf A-fragment (16-bit A 16x32 layout) from two 16B runs.
__device__ __forceinline__ v16bf load_afrag(const __bf16* rowPtr, int k0, int g0) {
    v8bf lo = *(const v8bf*)(rowPtr + k0 + g0);        // K = g0 .. g0+7
    v8bf hi = *(const v8bf*)(rowPtr + k0 + g0 + 16);   // K = g0+16 .. g0+23
    v16bf a;
    #pragma unroll
    for (int j = 0; j < 8; ++j) { a[j] = lo[j]; a[8 + j] = hi[j]; }
    return a;
}

// ---------------------------------------------------------------------------
// Router: logits^T = W^T(16x32 A) x x^T(32x16 B). Each wave owns 16 tokens
// and all 64 experts for BOTH projections (8 WMMA chains). No LDS/barriers in
// the K loop. Issue order per chunk: weight loads (L2) FIRST, next-chunk x
// loads (HBM) LAST, so in-order LOADcnt waits for the WMMAs never force the
// long-latency x loads to complete early.
// ---------------------------------------------------------------------------
__global__ void __launch_bounds__(kWavesPerBlk * 32) router_kernel(
    const float*  __restrict__ x,        // [Rows][D] fp32
    const __bf16* __restrict__ WgT,      // [E][D] bf16
    const __bf16* __restrict__ WnT,      // [E][D] bf16
    const float*  __restrict__ gnoise,   // [Rows][E] fp32
    float*        __restrict__ probs_out,  // [Rows][8]
    int*          __restrict__ idx_out)    // [Rows][8]
{
    __shared__ float ldsN[kTokPerBlk][kE + 1];   // +1 pad: conflict-free scans

    const int tid  = threadIdx.x;
    const int lane = tid & 31;
    const int wave = tid >> 5;

    // B (x^T) mapping: column N = token = lane&15; lanes 0-15 hold K {0..7},{16..23},
    // lanes 16-31 hold K {8..15},{24..31} -> two 32B fp32 runs per lane per chunk.
    const int tcol  = lane & 15;
    const int khB   = (lane >> 4) * 8;
    const int tokBase = blockIdx.x * kTokPerBlk + wave * kTokPerWave;
    const float* xRow = x + (size_t)(tokBase + tcol) * kD + khB;

    // A (W^T) mapping: row M = expert = tile*16 + (lane&15);
    // lane<16 -> K groups {0..7},{16..23}; lane>=16 -> {8..15},{24..31}.
    const int aG0  = (lane < 16) ? 0 : 8;
    const int mOff = aG0;                            // C/D row offset too
    const __bf16* wgRow[4];
    const __bf16* wnRow[4];
    #pragma unroll
    for (int t = 0; t < 4; ++t) {
        const size_t er = (size_t)(16 * t + (lane & 15)) * kD;
        wgRow[t] = WgT + er;
        wnRow[t] = WnT + er;
    }

    v8f accG[4] = {{}, {}, {}, {}};
    v8f accN[4] = {{}, {}, {}, {}};

    // ---- prologue: issue first chunk's x loads ----
    v4f f0 = __builtin_nontemporal_load((const v4f*)(xRow + 0));
    v4f f1 = __builtin_nontemporal_load((const v4f*)(xRow + 4));
    v4f f2 = __builtin_nontemporal_load((const v4f*)(xRow + 16));
    v4f f3 = __builtin_nontemporal_load((const v4f*)(xRow + 20));

    for (int k0 = 0; k0 < kD; k0 += kTileK) {
        // ---- 1) ALL weight loads for this chunk (16x global_load_b128, L2) ----
        v16bf af[8];
        #pragma unroll
        for (int t = 0; t < 4; ++t) {
            af[2 * t]     = load_afrag(wgRow[t], k0, aG0);
            af[2 * t + 1] = load_afrag(wnRow[t], k0, aG0);
        }

        // ---- 2) NEXT chunk's x loads, issued last (wrap keeps in-bounds) ----
        const int kn = (k0 + kTileK) & (kD - 1);
        v4f n0 = __builtin_nontemporal_load((const v4f*)(xRow + kn));
        v4f n1 = __builtin_nontemporal_load((const v4f*)(xRow + kn + 4));
        v4f n2 = __builtin_nontemporal_load((const v4f*)(xRow + kn + 16));
        v4f n3 = __builtin_nontemporal_load((const v4f*)(xRow + kn + 20));
        __builtin_prefetch((const void*)(xRow + k0 + 8 * kTileK), 0, 0);

        // ---- 3) B fragment from x loads issued LAST iteration ----
        v16bf b;
        #pragma unroll
        for (int j = 0; j < 4; ++j) {
            b[j]      = (__bf16)f0[j];
            b[4 + j]  = (__bf16)f1[j];
            b[8 + j]  = (__bf16)f2[j];
            b[12 + j] = (__bf16)f3[j];
        }

        // ---- 4) 8 independent WMMAs (staggered weight-load waits only) ----
        #pragma unroll
        for (int t = 0; t < 4; ++t) {
            accG[t] = __builtin_amdgcn_wmma_f32_16x16x32_bf16(
                          false, af[2 * t], false, b, (short)0, accG[t], false, false);
            accN[t] = __builtin_amdgcn_wmma_f32_16x16x32_bf16(
                          false, af[2 * t + 1], false, b, (short)0, accN[t], false, false);
        }

        f0 = n0; f1 = n1; f2 = n2; f3 = n3;
    }

    // ---- epilogue: noisy = gate + softplus(noise) * gnoise ----
    // C/D layout: lane holds token column (lane&15); VGPR v holds expert row
    // 16*t + v (lanes 0-15) or 16*t + v + 8 (lanes 16-31).
    const int tokLocal = wave * kTokPerWave + tcol;
    const int grow     = blockIdx.x * kTokPerBlk + tokLocal;
    const float* gBase = gnoise + (size_t)grow * kE;
    #pragma unroll
    for (int t = 0; t < 4; ++t) {
        const int eBase = 16 * t + mOff;
        v4f g0 = __builtin_nontemporal_load((const v4f*)(gBase + eBase));
        v4f g1 = __builtin_nontemporal_load((const v4f*)(gBase + eBase + 4));
        #pragma unroll
        for (int v = 0; v < 8; ++v) {
            const float gn = (v < 4) ? g0[v] : g1[v - 4];
            const float nz = accN[t][v];
            // stable softplus with hardware exp/log
            const float sp = fmaxf(nz, 0.0f) + __logf(1.0f + __expf(-fabsf(nz)));
            ldsN[tokLocal][eBase + v] = accG[t][v] + sp * gn;
        }
    }
    __syncthreads();

    // ---- top-8 + softmax: one thread per token ----
    if (tid < kTokPerBlk) {
        float topv[kTopK];
        int   topi[kTopK];
        #pragma unroll
        for (int i = 0; i < kTopK; ++i) {
            float best = -INFINITY;
            int   bidx = 0;
            for (int j = 0; j < kE; ++j) {
                float tv = ldsN[tid][j];
                if (tv > best) { best = tv; bidx = j; }  // strict > => lowest index on ties
            }
            topv[i] = best;
            topi[i] = bidx;
            ldsN[tid][bidx] = -INFINITY;
        }
        const float mx = topv[0];
        float ex[kTopK];
        float sum = 0.0f;
        #pragma unroll
        for (int i = 0; i < kTopK; ++i) { ex[i] = __expf(topv[i] - mx); sum += ex[i]; }
        const float inv = 1.0f / sum;

        const int orow = blockIdx.x * kTokPerBlk + tid;
        #pragma unroll
        for (int i = 0; i < kTopK; ++i) {
            probs_out[(size_t)orow * kTopK + i] = ex[i] * inv;
            idx_out  [(size_t)orow * kTopK + i] = topi[i];
        }
    }
}

// ---------------------------------------------------------------------------
extern "C" void kernel_launch(void* const* d_in, const int* in_sizes, int n_in,
                              void* d_out, int out_size, void* d_ws, size_t ws_size,
                              hipStream_t stream) {
    const float* x       = (const float*)d_in[0];   // [4,4096,4096]
    const float* W_gate  = (const float*)d_in[1];   // [4096,64]
    const float* W_noise = (const float*)d_in[2];   // [4096,64]
    const float* gnoise  = (const float*)d_in[3];   // [4,4096,64]

    // workspace: transposed bf16 weights (2 * 64*4096*2B = 1 MB)
    __bf16* WgT = (__bf16*)d_ws;
    __bf16* WnT = WgT + (size_t)kD * kE;

    // output tuple: probs fp32 [Rows][8], then indices int32 [Rows][8]
    float* probs = (float*)d_out;
    int*   idx   = (int*)((float*)d_out + (size_t)kRows * kTopK);

    cvt_weights_kernel<<<(kD * kE + 255) / 256, 256, 0, stream>>>(
        W_gate, W_noise, WgT, WnT);

    router_kernel<<<kRows / kTokPerBlk, kWavesPerBlk * 32, 0, stream>>>(
        x, WgT, WnT, gnoise, probs, idx);
}
